// AttDecode_66735201845258
// MI455X (gfx1250) — matrix operand order repaired and checked by reading, not compiled, not run on hardware
//
#include <hip/hip_runtime.h>
#include <math.h>
#include <stdint.h>

#define BATCHN 16
#define SEQ    512
#define EMB    200
#define VOCAB  32000
#define NHEAD  8
#define LN_EPS 1e-5f
#define KP     224   // EMB padded to k-step multiple (32)
#define NPE    256   // padded N for EMB-output weights (multiple of 128)

typedef __attribute__((ext_vector_type(16))) __bf16 v16bf;
typedef __attribute__((ext_vector_type(8)))  __bf16 v8bf;
typedef __attribute__((ext_vector_type(8)))  float  v8f;

__device__ __forceinline__ __bf16 f2bf(float f) {
  unsigned u = __builtin_bit_cast(unsigned, f);
  u += 0x7FFFu + ((u >> 16) & 1u);                 // round-to-nearest-even
  unsigned short s = (unsigned short)(u >> 16);
  return __builtin_bit_cast(__bf16, s);
}
__device__ __forceinline__ unsigned short f2bfu(float f) {
  unsigned u = __builtin_bit_cast(unsigned, f);
  u += 0x7FFFu + ((u >> 16) & 1u);
  return (unsigned short)(u >> 16);
}

// ---------------------------------------------------------------------------
// h[b,s,0:200] = embed[y[b,s],:] + pos[s,:]   (row stride KP)
// ---------------------------------------------------------------------------
__global__ void embed_add_kernel(const int* __restrict__ y,
                                 const float* __restrict__ embed,
                                 const float* __restrict__ pos,
                                 float* __restrict__ h) {
  const int row = blockIdx.x;               // b*SEQ + s
  const int s   = row & (SEQ - 1);
  const int v   = y[row];
  const float* e = embed + (long)v * EMB;
  const float* p = pos   + (long)s * EMB;
  float* o = h + (long)row * KP;
  for (int i = threadIdx.x; i < EMB; i += blockDim.x) o[i] = e[i] + p[i];
}

// Zero the pad columns [200,224) of a padded activation buffer.
__global__ void zero_pad_kernel(float* __restrict__ buf) {
  const int row = blockIdx.x, t = threadIdx.x;
  if (t < KP - EMB) buf[(long)row * KP + EMB + t] = 0.f;
}

// xp[row, 0:224] = {x[row, 0:200], zeros}
__global__ void pad_copy_kernel(const float* __restrict__ x, float* __restrict__ xp) {
  const int row = blockIdx.x;
  for (int i = threadIdx.x; i < KP; i += blockDim.x)
    xp[(long)row * KP + i] = (i < EMB) ? x[(long)row * EMB + i] : 0.f;
}

// ---------------------------------------------------------------------------
// fp32 [K,N] weight -> bf16 transposed [Npad][Kpad] (zero-filled padding).
// 32x32 LDS tile transpose. blockDim 256. grid (Npad/32, Kpad/32).
// ---------------------------------------------------------------------------
__global__ __launch_bounds__(256) void conv_wT_bf16(
    const float* __restrict__ W, __bf16* __restrict__ out,
    int K, int N, int Kpad, int Npad) {
  __shared__ __bf16 tile[32][33];
  const int kb = blockIdx.y * 32, nb = blockIdx.x * 32;
  const int tx = threadIdx.x & 31, ty = threadIdx.x >> 5;
#pragma unroll
  for (int i = 0; i < 4; ++i) {
    int k = kb + ty + 8 * i, n = nb + tx;
    float v = (k < K && n < N) ? W[(long)k * N + n] : 0.f;
    tile[ty + 8 * i][tx] = f2bf(v);
  }
  __syncthreads();
#pragma unroll
  for (int i = 0; i < 4; ++i) {
    int n = nb + ty + 8 * i, k = kb + tx;
    if (n < Npad && k < Kpad) out[(long)n * Kpad + k] = tile[tx][ty + 8 * i];
  }
}

// WfuseSum^T[n][k] = sum_h Wfuse[h*200+k, n]   (tile(c,HEAD)@Wfuse == c@WfuseSum)
__global__ void fuse_sumT_bf16(const float* __restrict__ Wfuse,
                               __bf16* __restrict__ out, long total) {
  long idx = (long)blockIdx.x * blockDim.x + threadIdx.x;
  if (idx >= total) return;
  int n = (int)(idx / KP), k = (int)(idx % KP);
  float s = 0.f;
  if (n < EMB && k < 200) {
#pragma unroll
    for (int hh = 0; hh < NHEAD; ++hh)
      s += Wfuse[(long)(hh * 200 + k) * EMB + n];
  }
  out[idx] = f2bf(s);
}

// ---------------------------------------------------------------------------
// WMMA bf16 GEMM:  C = scale * (A @ B) + bias  [+ ReLU]
//   A: fp32 [M rows, lda stride], K multiple of 32, all K cols readable
//      (padded with zeros) -> staging has NO bounds checks.
//   B: if B_BF16:  bf16 [Npad][ldb=Kpad] pre-transposed, zero-padded;
//                  staged via global_load_async_to_lds_b128 (ASYNCcnt)
//      elif B_TRANS: fp32, B[k,n] = Bf[n*ldb + k] (padded rows; check-free)
//      else:         fp32, B[k,n] = Bf[k*ldb + n], guarded by gn < Nbuf
//   C: fp32 [M, ldc], written only for gn < N (pads untouched).
// Block 256 thr (8 wave32). Tile M=128 (16 rows/wave), N=128, K-step 32.
// Batched via gridDim.z with strides sA/sB/sC (weights: sB=0).
// ---------------------------------------------------------------------------
template <bool B_BF16, bool B_TRANS, bool DO_RELU>
__global__ __launch_bounds__(256) void gemm_wmma(
    const float* __restrict__ A, const void* __restrict__ Bp,
    const float* __restrict__ bias, float* __restrict__ C,
    int M, int N, int K, int lda, int ldb, int ldc, int Nbuf,
    long sA, long sB, long sC, float scale) {
  constexpr int BM = 128, BN = 128, BK = 32;
  __shared__ __bf16 Asl[BM][BK];
  __shared__ __bf16 Bsl[BN][BK];

  const int z = blockIdx.z;
  A += (long)z * sA;
  C += (long)z * sC;
  const float*  Bf = (const float*)Bp + (long)z * sB;
  const __bf16* Bh = (const __bf16*)Bp;

  const int m0 = blockIdx.y * BM, n0 = blockIdx.x * BN;
  const int tid = threadIdx.x, lane = tid & 31, w = tid >> 5;

  v8f acc[BN / 16];
  v8f zero = {0.f, 0.f, 0.f, 0.f, 0.f, 0.f, 0.f, 0.f};
#pragma unroll
  for (int t = 0; t < BN / 16; ++t) acc[t] = zero;

  for (int k0 = 0; k0 < K; k0 += BK) {
    // ---- stage B tile into Bsl[n][k] ----
    if constexpr (B_BF16) {
      // 512 async 16B transfers (2/lane): 128 rows x 64B, contiguous source.
#pragma unroll
      for (int u = tid; u < BN * 4; u += 256) {
        const int n = u >> 2, koff = (u & 3) * 8;
        const __bf16* src = Bh + (long)(n0 + n) * ldb + (k0 + koff);
        unsigned loff = (unsigned)(uintptr_t)&Bsl[n][koff];  // flat->LDS offset
        asm volatile("global_load_async_to_lds_b128 %0, %1, off"
                     :: "v"(loff), "v"((unsigned long long)(uintptr_t)src)
                     : "memory");
      }
    } else if constexpr (B_TRANS) {
      // B[k][n] = Bf[n*ldb + k]: contiguous in k, padded rows -> check-free.
#pragma unroll
      for (int u = tid; u < BN * BK / 4; u += 256) {
        int n = u >> 3, c4 = (u & 7) * 4;
        const float* src = Bf + (long)(n0 + n) * ldb + (k0 + c4);
        float4 f = *(const float4*)src;
        ushort4 p = {f2bfu(f.x), f2bfu(f.y), f2bfu(f.z), f2bfu(f.w)};
        *(ushort4*)&Bsl[n][c4] = p;
      }
    } else {
      // B[k][n] = Bf[k*ldb + n]: transpose-stage; gn guarded by Nbuf.
#pragma unroll
      for (int u = tid; u < (BN / 4) * BK; u += 256) {
        int kk = u >> 5, n4 = (u & 31) * 4;
        int gn = n0 + n4;
        float4 f = {0.f, 0.f, 0.f, 0.f};
        if (gn < Nbuf) f = *(const float4*)(Bf + (long)(k0 + kk) * ldb + gn);
        Bsl[n4 + 0][kk] = f2bf(f.x);
        Bsl[n4 + 1][kk] = f2bf(f.y);
        Bsl[n4 + 2][kk] = f2bf(f.z);
        Bsl[n4 + 3][kk] = f2bf(f.w);
      }
    }

    // ---- stage A tile (fp32 -> bf16): 1024 float4 units, 4/thread ----
#pragma unroll
    for (int u = tid; u < BM * BK / 4; u += 256) {
      int r = u >> 3, c4 = (u & 7) * 4;
      const float* src = A + (long)(m0 + r) * lda + (k0 + c4);
      float4 f = *(const float4*)src;
      __builtin_prefetch(src + BK, 0, 0);          // next k-tile (speculative)
      ushort4 p = {f2bfu(f.x), f2bfu(f.y), f2bfu(f.z), f2bfu(f.w)};
      *(ushort4*)&Asl[r][c4] = p;
    }

    if constexpr (B_BF16)
      asm volatile("s_wait_asynccnt 0" ::: "memory");
    __syncthreads();

    // ---- A fragment: row = w*16 + lane%16; K halves split per lane group ----
    const int mrow = w * 16 + (lane & 15);
    const int ka = (lane >> 4) * 8;  // lanes 0-15: K 0..7 & 16..23; 16-31: 8..15 & 24..31
    v8bf alo = *(const v8bf*)&Asl[mrow][ka];
    v8bf ahi = *(const v8bf*)&Asl[mrow][ka + 16];
    v16bf afrag;
#pragma unroll
    for (int e = 0; e < 8; ++e) { afrag[e] = alo[e]; afrag[8 + e] = ahi[e]; }

    // ---- B fragments + WMMA (8 n-subtiles per wave) ----
    const int kb = (lane >> 4) * 8;  // lanes 0-15: K 0..7 & 16..23; 16-31: 8..15 & 24..31
#pragma unroll
    for (int t = 0; t < BN / 16; ++t) {
      const int ncol = t * 16 + (lane & 15);
      v8bf blo = *(const v8bf*)&Bsl[ncol][kb];
      v8bf bhi = *(const v8bf*)&Bsl[ncol][kb + 16];
      v16bf bfrag;
#pragma unroll
      for (int e = 0; e < 8; ++e) { bfrag[e] = blo[e]; bfrag[8 + e] = bhi[e]; }
      acc[t] = __builtin_amdgcn_wmma_f32_16x16x32_bf16(
          false, afrag, false, bfrag, (short)0, acc[t], false, false);
    }
    __syncthreads();
  }

  // ---- epilogue: D layout — lane%16 = N col; rows (lane>=16)*8 + vgpr ----
#pragma unroll
  for (int t = 0; t < BN / 16; ++t) {
    const int gn = n0 + t * 16 + (lane & 15);
#pragma unroll
    for (int r = 0; r < 8; ++r) {
      const int gm = m0 + w * 16 + (lane >> 4) * 8 + r;
      if (gn < N) {
        float v = acc[t][r] * scale;
        if (bias) v += bias[gn];
        if (DO_RELU) v = fmaxf(v, 0.f);
        C[(long)gm * (long)ldc + gn] = v;
      }
    }
  }
}

// ---------------------------------------------------------------------------
// Row softmax over 512 columns; one block (256 threads) per row, in place.
// ---------------------------------------------------------------------------
__global__ __launch_bounds__(256) void softmax512(float* __restrict__ s) {
  __shared__ float red[256];
  float* p = s + (long)blockIdx.x * SEQ;
  const int t = threadIdx.x;
  float a = p[t], b = p[t + 256];
  red[t] = fmaxf(a, b);
  __syncthreads();
  for (int o = 128; o > 0; o >>= 1) {
    if (t < o) red[t] = fmaxf(red[t], red[t + o]);
    __syncthreads();
  }
  const float m = red[0];
  __syncthreads();
  float ea = __expf(a - m), eb = __expf(b - m);
  red[t] = ea + eb;
  __syncthreads();
  for (int o = 128; o > 0; o >>= 1) {
    if (t < o) red[t] += red[t + o];
    __syncthreads();
  }
  const float inv = 1.f / red[0];
  p[t] = ea * inv;
  p[t + 256] = eb * inv;
}

// ---------------------------------------------------------------------------
// h = LayerNorm_{(S,E) joint}(z + h); one block per batch; padded layout (KP).
// ---------------------------------------------------------------------------
__global__ __launch_bounds__(256) void add_ln_kernel(const float* __restrict__ z,
                                                     float* __restrict__ h) {
  __shared__ float rs[256], rq[256];
  const long base = (long)blockIdx.x * SEQ * KP;
  const int t = threadIdx.x;
  const int n = SEQ * EMB;
  float s = 0.f, q = 0.f;
  for (int idx = t; idx < n; idx += 256) {
    int r = idx / EMB, c = idx - r * EMB;
    long o = base + (long)r * KP + c;
    float v = z[o] + h[o];
    s += v;
    q += v * v;
  }
  rs[t] = s; rq[t] = q;
  __syncthreads();
  for (int o = 128; o > 0; o >>= 1) {
    if (t < o) { rs[t] += rs[t + o]; rq[t] += rq[t + o]; }
    __syncthreads();
  }
  const float mean = rs[0] / n;
  const float var  = rq[0] / n - mean * mean;
  const float rstd = rsqrtf(var + LN_EPS);
  for (int idx = t; idx < n; idx += 256) {
    int r = idx / EMB, c = idx - r * EMB;
    long o = base + (long)r * KP + c;
    float v = z[o] + h[o];
    h[o] = (v - mean) * rstd;
  }
}

// ---------------------------------------------------------------------------
extern "C" void kernel_launch(void* const* d_in, const int* in_sizes, int n_in,
                              void* d_out, int out_size, void* d_ws, size_t ws_size,
                              hipStream_t stream) {
  (void)in_sizes; (void)n_in; (void)out_size; (void)ws_size;
  const float* x     = (const float*)d_in[0];
  const int*   y     = (const int*)d_in[1];
  const float* embed = (const float*)d_in[2];
  const float* pos   = (const float*)d_in[3];
  const float* Wqkv  = (const float*)d_in[4];
  const float* bqkv  = (const float*)d_in[5];
  const float* Wfuse = (const float*)d_in[6];
  const float* bfuse = (const float*)d_in[7];
  const float* W1    = (const float*)d_in[8];
  const float* b1    = (const float*)d_in[9];
  const float* W2    = (const float*)d_in[10];
  const float* b2    = (const float*)d_in[11];
  const float* Wout  = (const float*)d_in[12];
  const float* bout  = (const float*)d_in[13];
  float* out = (float*)d_out;

  const int M = BATCHN * SEQ;  // 8192

  size_t off = 0;
  auto alloc = [&](size_t bytes) -> void* {
    off = (off + 255) & ~(size_t)255;
    void* p = (char*)d_ws + off;
    off += bytes;
    return p;
  };
  float*  h      = (float*)alloc((size_t)M * KP * 4);
  float*  qb     = (float*)alloc((size_t)M * KP * 4);
  float*  kvx    = (float*)alloc((size_t)M * KP * 4);
  float*  cb     = (float*)alloc((size_t)M * KP * 4);
  float*  t1     = (float*)alloc((size_t)M * KP * 4);
  float*  t2     = (float*)alloc((size_t)M * KP * 4);
  float*  xp     = (float*)alloc((size_t)M * KP * 4);
  float*  sc     = (float*)alloc((size_t)BATCHN * SEQ * SEQ * 4);
  __bf16* wqkv_b = (__bf16*)alloc((size_t)NPE * KP * 2);
  __bf16* wfs_b  = (__bf16*)alloc((size_t)NPE * KP * 2);
  __bf16* w1_b   = (__bf16*)alloc((size_t)NPE * KP * 2);
  __bf16* w2_b   = (__bf16*)alloc((size_t)NPE * KP * 2);
  __bf16* wout_b = (__bf16*)alloc((size_t)VOCAB * KP * 2);

  // Zero pad columns of all padded activation buffers; copy-pad x.
  zero_pad_kernel<<<M, 32, 0, stream>>>(h);
  zero_pad_kernel<<<M, 32, 0, stream>>>(qb);
  zero_pad_kernel<<<M, 32, 0, stream>>>(kvx);
  zero_pad_kernel<<<M, 32, 0, stream>>>(cb);
  zero_pad_kernel<<<M, 32, 0, stream>>>(t1);
  zero_pad_kernel<<<M, 32, 0, stream>>>(t2);
  pad_copy_kernel<<<M, 256, 0, stream>>>(x, xp);

  // h = embed[y] + pos
  embed_add_kernel<<<M, 256, 0, stream>>>(y, embed, pos, h);

  // Weight conversion: fp32 [K,N] -> bf16 transposed [Npad][Kpad]
  {
    dim3 ge(NPE / 32, KP / 32);       // (8, 7)
    conv_wT_bf16<<<ge, 256, 0, stream>>>(Wqkv, wqkv_b, EMB, EMB, KP, NPE);
    conv_wT_bf16<<<ge, 256, 0, stream>>>(W1, w1_b, EMB, EMB, KP, NPE);
    conv_wT_bf16<<<ge, 256, 0, stream>>>(W2, w2_b, EMB, EMB, KP, NPE);
    long totf = (long)NPE * KP;
    fuse_sumT_bf16<<<(int)((totf + 255) / 256), 256, 0, stream>>>(Wfuse, wfs_b, totf);
    dim3 gv(VOCAB / 32, KP / 32);     // (1000, 7)
    conv_wT_bf16<<<gv, 256, 0, stream>>>(Wout, wout_b, EMB, VOCAB, KP, VOCAB);
  }

  const float inv_sqrt = 1.0f / sqrtf(200.0f);
  dim3 blk(256);

  // Activation @ bf16-weight GEMM: lda = ldc = KP (except final: ldc = VOCAB)
  auto gemm_w = [&](const float* A, const __bf16* Wb, const float* bias,
                    float* C, int N, int ldc, bool relu) {
    dim3 grid((N + 127) / 128, M / 128, 1);
    if (relu)
      gemm_wmma<true, false, true><<<grid, blk, 0, stream>>>(
          A, Wb, bias, C, M, N, KP, KP, KP, ldc, 0, 0, 0, 0, 1.0f);
    else
      gemm_wmma<true, false, false><<<grid, blk, 0, stream>>>(
          A, Wb, bias, C, M, N, KP, KP, KP, ldc, 0, 0, 0, 0, 1.0f);
  };

  auto attention = [&](const float* kv) {
    // scores = (q @ kv^T) / sqrt(200)  [batched 16x: 512x512, K=224 padded]
    dim3 gs(SEQ / 128, SEQ / 128, BATCHN);
    gemm_wmma<false, true, false><<<gs, blk, 0, stream>>>(
        qb, kv, nullptr, sc, SEQ, SEQ, KP, KP, KP, SEQ, 0,
        (long)SEQ * KP, (long)SEQ * KP, (long)SEQ * SEQ, inv_sqrt);
    softmax512<<<BATCHN * SEQ, 256, 0, stream>>>(sc);
    // c = a @ kv                       [batched 16x: 512x200, K=512]
    dim3 gc((EMB + 127) / 128, SEQ / 128, BATCHN);
    gemm_wmma<false, false, false><<<gc, blk, 0, stream>>>(
        sc, kv, nullptr, cb, SEQ, EMB, SEQ, SEQ, KP, KP, KP,
        (long)SEQ * SEQ, (long)SEQ * KP, (long)SEQ * KP, 1.0f);
    // tile(c,HEAD) @ Wfuse + bfuse  ==  c @ WfuseSum + bfuse
    gemm_w(cb, wfs_b, bfuse, t1, EMB, KP, false);
    add_ln_kernel<<<BATCHN, 256, 0, stream>>>(t1, h);
  };

  // Cross-attention kv = x @ Wqkv + bqkv is iteration-invariant
  gemm_w(xp, wqkv_b, bqkv, kvx, EMB, KP, false);

  for (int it = 0; it < 2; ++it) {
    // self attention: q == kv == h @ Wqkv + b (shared Linear, same input)
    gemm_w(h, wqkv_b, bqkv, qb, EMB, KP, false);
    attention(qb);
    // cross attention: q from h, kv from x (precomputed)
    gemm_w(h, wqkv_b, bqkv, qb, EMB, KP, false);
    attention(kvx);
    // feed forward
    gemm_w(h, w1_b, b1, t1, EMB, KP, true);
    gemm_w(t1, w2_b, b2, t2, EMB, KP, false);
    add_ln_kernel<<<BATCHN, 256, 0, stream>>>(t2, h);
  }

  // final projection: out = h @ Wout + bout   [8192 x 32000, K=224 padded]
  gemm_w(h, wout_b, bout, out, VOCAB, VOCAB, false);
}
